// TransformerDecoderLayer_83777632076508
// MI455X (gfx1250) — compile-verified, matrix operand order checked by reading, not run on hardware
//
#include <hip/hip_runtime.h>

// ---------------------------------------------------------------------------
// Problem constants (from the reference)
// ---------------------------------------------------------------------------
#define Bsz 32
#define Ssz 512
#define Dsz 1024
#define NHsz 16
#define Esz 16
#define Hsz 256
#define Ksel 8
#define Tsz (Bsz * Ssz)      // 16384 tokens
#define DH 64                // head dim
#define LN_EPS 1e-5f
#define W_MI 5e-4f

typedef __attribute__((ext_vector_type(16))) __bf16 v16bf;
typedef __attribute__((ext_vector_type(8)))  __bf16 v8bf;
typedef __attribute__((ext_vector_type(8)))  float  v8f;

// ---------------------------------------------------------------------------
// Helpers
// ---------------------------------------------------------------------------
__device__ __forceinline__ v8f wmma_bf16(v16bf a, v16bf b, v8f c) {
  return __builtin_amdgcn_wmma_f32_16x16x32_bf16(false, a, false, b, (short)0, c,
                                                 false, false);
}

// Load a 16x32 bf16 fragment (A or Bt operand) from a row-major [rows, ld]
// matrix. Lane layout per CDNA5 ISA: lanes 0-15 -> rows 0-15 / K 0..7,16..23;
// lanes 16-31 -> rows 0-15 / K 8..15,24..31.
__device__ __forceinline__ v16bf load_frag(const __bf16* base, int row0, int ld,
                                           int k0, int lane) {
  int g = (lane >> 4) & 1;
  int r = lane & 15;
  const __bf16* p = base + (size_t)(row0 + r) * ld + k0 + g * 8;
  v8bf lo = *(const v8bf*)(p);
  v8bf hi = *(const v8bf*)(p + 16);
  v16bf f;
#pragma unroll
  for (int i = 0; i < 8; ++i) { f[i] = lo[i]; f[i + 8] = hi[i]; }
  return f;
}

__device__ __forceinline__ float gelu_exact(float x) {
  return 0.5f * x * (1.0f + erff(x * 0.70710678118654752f));
}

// ---------------------------------------------------------------------------
// f32 -> bf16 convert (row-major copy)
// ---------------------------------------------------------------------------
__global__ void cvt_bf16_kernel(const float* __restrict__ in,
                                __bf16* __restrict__ out, int n) {
  int i = blockIdx.x * 256 + threadIdx.x;
  if (i < n) out[i] = (__bf16)in[i];
}

// out[b][c][r] = (bf16) in[b][r][c]   (batched transpose + convert)
__global__ void transpose_cvt_kernel(const float* __restrict__ in,
                                     __bf16* __restrict__ out, int Rin, int Cin,
                                     int n) {
  int i = blockIdx.x * 256 + threadIdx.x;
  if (i >= n) return;
  int r = i % Rin;
  int c = (i / Rin) % Cin;
  int b = i / (Rin * Cin);
  out[i] = (__bf16)in[(size_t)b * Rin * Cin + (size_t)r * Cin + c];
}

// Build V^T:  vt[(b*NH+h)*64*512 + d*512 + s] = qkv_bf16[(b*S+s)*3D + 2D + h*64 + d]
__global__ void vt_kernel(const __bf16* __restrict__ qkv,
                          __bf16* __restrict__ vt) {
  int i = blockIdx.x * 256 + threadIdx.x;   // over B*NH*64*512 = 16.7M
  int s = i & 511;
  int d = (i >> 9) & 63;
  int bh = i >> 15;
  int h = bh & 15;
  int b = bh >> 4;
  vt[i] = qkv[((size_t)(b * Ssz + s)) * (3 * Dsz) + 2 * Dsz + h * DH + d];
}

// ---------------------------------------------------------------------------
// LayerNorm -> bf16 (one token per block, 256 threads, D = 1024)
// ---------------------------------------------------------------------------
__global__ void ln_bf16_kernel(const float* __restrict__ x,
                               const float* __restrict__ gamma,
                               const float* __restrict__ beta,
                               __bf16* __restrict__ out) {
  int t = blockIdx.x;
  int tid = threadIdx.x;
  const float* row = x + (size_t)t * Dsz;
  float v[4], s = 0.f, s2 = 0.f;
#pragma unroll
  for (int i = 0; i < 4; ++i) {
    v[i] = row[tid + i * 256];
    s += v[i];
    s2 += v[i] * v[i];
  }
#pragma unroll
  for (int m = 16; m >= 1; m >>= 1) {
    s += __shfl_xor(s, m, 32);
    s2 += __shfl_xor(s2, m, 32);
  }
  __shared__ float ws1[8], ws2[8];
  __shared__ float mean_s, rstd_s;
  int wave = tid >> 5, lane = tid & 31;
  if (lane == 0) { ws1[wave] = s; ws2[wave] = s2; }
  __syncthreads();
  if (tid == 0) {
    float a = 0.f, b = 0.f;
#pragma unroll
    for (int i = 0; i < 8; ++i) { a += ws1[i]; b += ws2[i]; }
    float mean = a * (1.0f / Dsz);
    float var = b * (1.0f / Dsz) - mean * mean;
    mean_s = mean;
    rstd_s = rsqrtf(var + LN_EPS);
  }
  __syncthreads();
  float mean = mean_s, rstd = rstd_s;
#pragma unroll
  for (int i = 0; i < 4; ++i) {
    int c = tid + i * 256;
    out[(size_t)t * Dsz + c] = (__bf16)((v[i] - mean) * rstd * gamma[c] + beta[c]);
  }
}

// ---------------------------------------------------------------------------
// Generic bf16 WMMA GEMM:  C[M,N] = A[M,K] @ Bt[N,K]^T (+bias) (+residual)
// 8 waves / block, each wave a 32x64 tile: 2 A-frags + 4 B-frags -> 8 WMMA
// per k-step (1.33 WMMA per 16B-pair fragment load).
// ---------------------------------------------------------------------------
__global__ void gemm_bf16_nt(const __bf16* __restrict__ A,
                             const __bf16* __restrict__ Bt,
                             const float* __restrict__ bias,
                             const float* __restrict__ residual,
                             float* __restrict__ Cf, __bf16* __restrict__ Cb,
                             int M, int N, int Kd) {
  int lane = threadIdx.x & 31;
  int wave = threadIdx.x >> 5;
  int mtiles = M >> 5;   // 32-row tiles
  int ntiles = N >> 6;   // 64-col tiles
  int wid = blockIdx.x * 8 + wave;
  if (wid >= mtiles * ntiles) return;
  int mt = wid % mtiles;   // consecutive waves share the same n0 -> B reuse
  int nt = wid / mtiles;
  int m0 = mt * 32, n0 = nt * 64;

  v8f acc[2][4];
#pragma unroll
  for (int i = 0; i < 2; ++i)
#pragma unroll
    for (int j = 0; j < 4; ++j) acc[i][j] = (v8f){};

  int r = lane & 15;
  for (int k0 = 0; k0 < Kd; k0 += 32) {
    // pull the next k-chunk toward the WGP while this step's WMMAs drain
    if (k0 + 32 < Kd) {
      __builtin_prefetch(A + (size_t)(m0 + r) * Kd + k0 + 32, 0, 3);
      __builtin_prefetch(Bt + (size_t)(n0 + r) * Kd + k0 + 32, 0, 3);
      __builtin_prefetch(Bt + (size_t)(n0 + 32 + r) * Kd + k0 + 32, 0, 3);
    }
    v16bf a0 = load_frag(A, m0, Kd, k0, lane);
    v16bf a1 = load_frag(A, m0 + 16, Kd, k0, lane);
    v16bf b[4];
#pragma unroll
    for (int j = 0; j < 4; ++j) b[j] = load_frag(Bt, n0 + j * 16, Kd, k0, lane);
#pragma unroll
    for (int j = 0; j < 4; ++j) {
      acc[0][j] = wmma_bf16(a0, b[j], acc[0][j]);
      acc[1][j] = wmma_bf16(a1, b[j], acc[1][j]);
    }
  }

  int g = lane >> 4;
  int nl = lane & 15;
#pragma unroll
  for (int i = 0; i < 2; ++i) {
#pragma unroll
    for (int j = 0; j < 4; ++j) {
      int col = n0 + j * 16 + nl;
      float bv = bias ? bias[col] : 0.f;
#pragma unroll
      for (int v = 0; v < 8; ++v) {
        int row = m0 + i * 16 + v + 8 * g;
        float val = acc[i][j][v] + bv;
        if (residual) val += residual[(size_t)row * N + col];
        if (Cf) Cf[(size_t)row * N + col] = val;
        if (Cb) Cb[(size_t)row * N + col] = (__bf16)val;
      }
    }
  }
}

// ---------------------------------------------------------------------------
// Attention: one (b, h, q-tile of 16) per block. 256 threads = 8 waves.
// Phase 1: scores via WMMA -> LDS (f32). Phase 2: row softmax -> bf16 P in
// LDS.  Phase 3: ctx = P @ V via WMMA (A from LDS, B from global V^T).
// ---------------------------------------------------------------------------
__global__ void attn_kernel(const __bf16* __restrict__ qkv,
                            const __bf16* __restrict__ vt,
                            __bf16* __restrict__ ctx) {
  __shared__ float sc[16][516];       // 33 KB
  __shared__ __bf16 P[16][520];       // 16.6 KB

  int blk = blockIdx.x;               // B*NH*(S/16) = 16384
  int qt = blk & 31;
  int h = (blk >> 5) & 15;
  int b = blk >> 9;
  int wave = threadIdx.x >> 5, lane = threadIdx.x & 31;
  int g = lane >> 4, nl = lane & 15;

  int t0 = b * Ssz + qt * 16;
  const __bf16* qbase = qkv + (size_t)t0 * (3 * Dsz) + h * DH;
  const __bf16* kbase = qkv + (size_t)b * Ssz * (3 * Dsz) + Dsz + h * DH;

  // Q fragments (dh = 64 -> two k-steps), hoisted out of the key loop.
  v16bf qa0 = load_frag(qbase, 0, 3 * Dsz, 0, lane);
  v16bf qa1 = load_frag(qbase, 0, 3 * Dsz, 32, lane);

  // --- phase 1: scores ---
  for (int kt = wave * 4; kt < wave * 4 + 4; ++kt) {
    v8f acc = {};
    v16bf kb0 = load_frag(kbase, kt * 16, 3 * Dsz, 0, lane);
    v16bf kb1 = load_frag(kbase, kt * 16, 3 * Dsz, 32, lane);
    acc = wmma_bf16(qa0, kb0, acc);
    acc = wmma_bf16(qa1, kb1, acc);
    int col = kt * 16 + nl;
#pragma unroll
    for (int v = 0; v < 8; ++v) sc[v + 8 * g][col] = acc[v] * 0.125f;
  }
  __syncthreads();

  // --- phase 2: softmax (2 rows per wave) ---
  for (int r = wave * 2; r < wave * 2 + 2; ++r) {
    float mx = -1e30f;
    for (int c = lane; c < Ssz; c += 32) mx = fmaxf(mx, sc[r][c]);
#pragma unroll
    for (int m = 16; m >= 1; m >>= 1) mx = fmaxf(mx, __shfl_xor(mx, m, 32));
    float sum = 0.f;
    for (int c = lane; c < Ssz; c += 32) {
      float e = __expf(sc[r][c] - mx);
      sc[r][c] = e;
      sum += e;
    }
#pragma unroll
    for (int m = 16; m >= 1; m >>= 1) sum += __shfl_xor(sum, m, 32);
    float inv = 1.0f / sum;
    for (int c = lane; c < Ssz; c += 32) P[r][c] = (__bf16)(sc[r][c] * inv);
  }
  __syncthreads();

  // --- phase 3: ctx = P @ V  (waves 0..3, one 16x16 d-tile each) ---
  if (wave < 4) {
    int n0 = wave * 16;
    const __bf16* vbase = vt + ((size_t)(b * NHsz + h) * DH + n0) * Ssz;
    v8f acc = {};
    for (int k0 = 0; k0 < Ssz; k0 += 32) {
      v16bf pa = load_frag(&P[0][0], 0, 520, k0, lane);
      v16bf vb = load_frag(vbase, 0, Ssz, k0, lane);
      acc = wmma_bf16(pa, vb, acc);
    }
    int col = h * DH + n0 + nl;
#pragma unroll
    for (int v = 0; v < 8; ++v)
      ctx[(size_t)(t0 + v + 8 * g) * Dsz + col] = (__bf16)acc[v];
  }
}

// ---------------------------------------------------------------------------
// Gate: per-token logits, softmax, rank-based top-8, writes probs + dense gates
// ---------------------------------------------------------------------------
__global__ void gate_kernel(const __bf16* __restrict__ h3,
                            const float* __restrict__ gate_w,
                            const int* __restrict__ task_id,
                            float* __restrict__ probs,
                            float* __restrict__ gd) {
  int t = blockIdx.x, tid = threadIdx.x;
  const float* gw = gate_w + (size_t)(*task_id) * Dsz * Esz;
  float acc[Esz];
#pragma unroll
  for (int e = 0; e < Esz; ++e) acc[e] = 0.f;
#pragma unroll
  for (int i = 0; i < 4; ++i) {
    int d = tid + i * 256;
    float hv = (float)h3[(size_t)t * Dsz + d];
    const float* gr = gw + (size_t)d * Esz;
#pragma unroll
    for (int e = 0; e < Esz; ++e) acc[e] += hv * gr[e];
  }
#pragma unroll
  for (int m = 16; m >= 1; m >>= 1)
#pragma unroll
    for (int e = 0; e < Esz; ++e) acc[e] += __shfl_xor(acc[e], m, 32);

  __shared__ float part[8][Esz];
  int wave = tid >> 5, lane = tid & 31;
  if (lane == 0)
#pragma unroll
    for (int e = 0; e < Esz; ++e) part[wave][e] = acc[e];
  __syncthreads();
  if (tid == 0) {
    float lg[Esz], p[Esz];
    for (int e = 0; e < Esz; ++e) {
      float s = 0.f;
      for (int w = 0; w < 8; ++w) s += part[w][e];
      lg[e] = s;
    }
    float mx = lg[0];
    for (int e = 1; e < Esz; ++e) mx = fmaxf(mx, lg[e]);
    float sum = 0.f;
    for (int e = 0; e < Esz; ++e) { p[e] = __expf(lg[e] - mx); sum += p[e]; }
    float inv = 1.0f / sum;
    for (int e = 0; e < Esz; ++e) p[e] *= inv;
    // top-8 (ties broken by lower index, matching jax.lax.top_k)
    float selsum = 0.f;
    bool sel[Esz];
    for (int e = 0; e < Esz; ++e) {
      int rank = 0;
      for (int j = 0; j < Esz; ++j)
        rank += (p[j] > p[e]) || (p[j] == p[e] && j < e);
      sel[e] = rank < Ksel;
      if (sel[e]) selsum += p[e];
    }
    float innorm = 1.0f / (selsum + 1e-6f);
    for (int e = 0; e < Esz; ++e) {
      probs[(size_t)t * Esz + e] = p[e];
      gd[(size_t)t * Esz + e] = sel[e] ? p[e] * innorm : 0.f;
    }
  }
}

// mean over tokens of probs (deterministic tree reduce, one expert per block)
__global__ void meanp_kernel(const float* __restrict__ probs,
                             float* __restrict__ mp) {
  int e = blockIdx.x, tid = threadIdx.x;
  float s = 0.f;
  for (int t = tid; t < Tsz; t += 256) s += probs[(size_t)t * Esz + e];
#pragma unroll
  for (int m = 16; m >= 1; m >>= 1) s += __shfl_xor(s, m, 32);
  __shared__ float part[8];
  if ((tid & 31) == 0) part[tid >> 5] = s;
  __syncthreads();
  if (tid == 0) {
    float a = 0.f;
    for (int i = 0; i < 8; ++i) a += part[i];
    mp[e] = a * (1.0f / Tsz);
  }
}

__global__ void aux_kernel(const float* __restrict__ mp,
                           float* __restrict__ out_aux) {
  int lane = threadIdx.x;
  float term = 0.f;
  if (lane < Esz) {
    float v = mp[lane];
    term = v * __logf(v + 1e-6f);
  }
#pragma unroll
  for (int m = 16; m >= 1; m >>= 1) term += __shfl_xor(term, m, 32);
  if (lane == 0) *out_aux = W_MI * term;
}

// x[t,:] += gate_dense[t,:] @ b2
__global__ void gbias_kernel(const float* __restrict__ gd,
                             const float* __restrict__ b2,
                             float* __restrict__ x) {
  int t = blockIdx.x, tid = threadIdx.x;
  __shared__ float gr[Esz];
  if (tid < Esz) gr[tid] = gd[(size_t)t * Esz + tid];
  __syncthreads();
#pragma unroll
  for (int i = 0; i < 4; ++i) {
    int d = tid + i * 256;
    float s = 0.f;
#pragma unroll
    for (int e = 0; e < Esz; ++e) s += gr[e] * b2[e * Dsz + d];
    x[(size_t)t * Dsz + d] += s;
  }
}

// ---------------------------------------------------------------------------
// MoE: 16-token tile per block.  For each expert: U = gelu(h3 @ w1t^T + b1)
// * gate (bf16 -> LDS), then out += U @ w2t^T.  h3 tile is cached in LDS.
// ---------------------------------------------------------------------------
__global__ void moe_kernel(const __bf16* __restrict__ h3,
                           const __bf16* __restrict__ w1t,
                           const float* __restrict__ b1,
                           const __bf16* __restrict__ w2t,
                           const float* __restrict__ gd,
                           float* __restrict__ x) {
  __shared__ __bf16 Hs[16][1032];   // 33 KB: h3 tile, ld padded for 16B align
  __shared__ __bf16 U[16][264];     // 8.4 KB: gated expert activations

  int t0 = blockIdx.x * 16;
  int tid = threadIdx.x;
  int wave = tid >> 5, lane = tid & 31;
  int g = lane >> 4, nl = lane & 15;

  // cooperative load of the h3 tile
  for (int i = tid; i < 16 * Dsz; i += 256) {
    int r = i >> 10, c = i & 1023;
    Hs[r][c] = h3[(size_t)(t0 + r) * Dsz + c];
  }
  __syncthreads();

  v8f out[8];
#pragma unroll
  for (int j = 0; j < 8; ++j) out[j] = (v8f){};

  for (int e = 0; e < Esz; ++e) {
    float gv[8];
#pragma unroll
    for (int v = 0; v < 8; ++v) gv[v] = gd[(size_t)(t0 + v + 8 * g) * Esz + e];

    // phase 1: U columns [wave*32, wave*32+32)
    v8f u0 = {}, u1 = {};
    const __bf16* w1b = w1t + (size_t)e * Hsz * Dsz;
    for (int k0 = 0; k0 < Dsz; k0 += 32) {
      v16bf a = load_frag(&Hs[0][0], 0, 1032, k0, lane);
      v16bf b0 = load_frag(w1b, wave * 32, Dsz, k0, lane);
      v16bf b1f = load_frag(w1b, wave * 32 + 16, Dsz, k0, lane);
      u0 = wmma_bf16(a, b0, u0);
      u1 = wmma_bf16(a, b1f, u1);
    }
    int c0 = wave * 32 + nl;
#pragma unroll
    for (int v = 0; v < 8; ++v) {
      float x0 = gelu_exact(u0[v] + b1[e * Hsz + c0]);
      float x1 = gelu_exact(u1[v] + b1[e * Hsz + c0 + 16]);
      U[v + 8 * g][c0] = (__bf16)(x0 * gv[v]);
      U[v + 8 * g][c0 + 16] = (__bf16)(x1 * gv[v]);
    }
    __syncthreads();

    // phase 2: out[:, wave*128 .. wave*128+128) += U @ w2t[e]^T
    const __bf16* w2b = w2t + (size_t)e * Dsz * Hsz;
    for (int k0 = 0; k0 < Hsz; k0 += 32) {
      v16bf a = load_frag(&U[0][0], 0, 264, k0, lane);
#pragma unroll
      for (int j = 0; j < 8; ++j) {
        v16bf bb = load_frag(w2b, wave * 128 + j * 16, Hsz, k0, lane);
        out[j] = wmma_bf16(a, bb, out[j]);
      }
    }
    __syncthreads();
  }

#pragma unroll
  for (int j = 0; j < 8; ++j) {
    int col = wave * 128 + j * 16 + nl;
#pragma unroll
    for (int v = 0; v < 8; ++v) {
      size_t idx = (size_t)(t0 + v + 8 * g) * Dsz + col;
      x[idx] += out[j][v];
    }
  }
}

// ---------------------------------------------------------------------------
// Host launcher
// ---------------------------------------------------------------------------
extern "C" void kernel_launch(void* const* d_in, const int* in_sizes, int n_in,
                              void* d_out, int out_size, void* d_ws,
                              size_t ws_size, hipStream_t stream) {
  const float* tgt        = (const float*)d_in[0];
  // d_in[1] = memory (unused in the norm_first path, faithful to reference)
  const float* ln1_g      = (const float*)d_in[2];
  const float* ln1_b      = (const float*)d_in[3];
  const float* ln3_g      = (const float*)d_in[4];
  const float* ln3_b      = (const float*)d_in[5];
  const float* in_proj_w  = (const float*)d_in[6];
  const float* in_proj_b  = (const float*)d_in[7];
  const float* out_proj_w = (const float*)d_in[8];
  const float* out_proj_b = (const float*)d_in[9];
  const float* gate_w     = (const float*)d_in[10];
  const float* w1         = (const float*)d_in[11];
  const float* b1         = (const float*)d_in[12];
  const float* w2         = (const float*)d_in[13];
  const float* b2         = (const float*)d_in[14];
  const int*   task_id    = (const int*)d_in[15];

  float* out_x    = (float*)d_out;                 // [T, D]
  float* out_aux  = out_x + (size_t)Tsz * Dsz;     // [1]
  float* out_prob = out_aux + 1;                   // [T, E]

  // workspace carve-up (bf16 buffers; regionX serves h1 -> ctx -> h3)
  char* wsb = (char*)d_ws;
  size_t off = 0;
  auto alloc = [&](size_t bytes) -> void* {
    void* p = wsb + off;
    off += (bytes + 255) & ~(size_t)255;
    return p;
  };
  __bf16* regionX    = (__bf16*)alloc((size_t)Tsz * Dsz * 2);       // 32 MB
  __bf16* qkv_bf     = (__bf16*)alloc((size_t)Tsz * 3 * Dsz * 2);   // 96 MB
  __bf16* vt_bf      = (__bf16*)alloc((size_t)Tsz * Dsz * 2);       // 32 MB
  __bf16* wq_bf      = (__bf16*)alloc((size_t)3 * Dsz * Dsz * 2);
  __bf16* wo_bf      = (__bf16*)alloc((size_t)Dsz * Dsz * 2);
  __bf16* w1t_bf     = (__bf16*)alloc((size_t)Esz * Hsz * Dsz * 2);
  __bf16* w2t_bf     = (__bf16*)alloc((size_t)Esz * Dsz * Hsz * 2);
  float*  gate_dense = (float*)alloc((size_t)Tsz * Esz * 4);
  float*  mp         = (float*)alloc(64);

  __bf16* h1_bf  = regionX;   // live: LN1 -> QKV gemm
  __bf16* ctx_bf = regionX;   // live: attn -> out-proj gemm
  __bf16* h3_bf  = regionX;   // live: LN3 -> gate/moe

  // 1) weight conversions
  {
    int n = 3 * Dsz * Dsz;
    cvt_bf16_kernel<<<(n + 255) / 256, 256, 0, stream>>>(in_proj_w, wq_bf, n);
  }
  {
    int n = Dsz * Dsz;
    cvt_bf16_kernel<<<(n + 255) / 256, 256, 0, stream>>>(out_proj_w, wo_bf, n);
  }
  {  // w1 [E][D][H] -> w1t [E][H][D]
    int n = Esz * Dsz * Hsz;
    transpose_cvt_kernel<<<(n + 255) / 256, 256, 0, stream>>>(w1, w1t_bf, Dsz,
                                                              Hsz, n);
  }
  {  // w2 [E][H][D] -> w2t [E][D][H]
    int n = Esz * Hsz * Dsz;
    transpose_cvt_kernel<<<(n + 255) / 256, 256, 0, stream>>>(w2, w2t_bf, Hsz,
                                                              Dsz, n);
  }

  // 2) LN1 -> h1 (bf16)
  ln_bf16_kernel<<<Tsz, 256, 0, stream>>>(tgt, ln1_g, ln1_b, h1_bf);

  // 3) QKV = h1 @ in_proj_w^T + b   ->  bf16 [T, 3D]
  {
    int waves = (Tsz / 32) * (3 * Dsz / 64);
    gemm_bf16_nt<<<waves / 8, 256, 0, stream>>>(h1_bf, wq_bf, in_proj_b,
                                                nullptr, nullptr, qkv_bf, Tsz,
                                                3 * Dsz, Dsz);
  }

  // 4) V^T for the ctx GEMM
  vt_kernel<<<(Tsz * Dsz) / 256, 256, 0, stream>>>(qkv_bf, vt_bf);

  // 5) attention -> ctx (bf16)
  attn_kernel<<<Bsz * NHsz * (Ssz / 16), 256, 0, stream>>>(qkv_bf, vt_bf,
                                                           ctx_bf);

  // 6) x = tgt + ctx @ out_proj_w^T + b   (f32 into d_out)
  {
    int waves = (Tsz / 32) * (Dsz / 64);
    gemm_bf16_nt<<<waves / 8, 256, 0, stream>>>(ctx_bf, wo_bf, out_proj_b, tgt,
                                                out_x, nullptr, Tsz, Dsz, Dsz);
  }

  // 7) LN3 -> h3 (bf16)
  ln_bf16_kernel<<<Tsz, 256, 0, stream>>>(out_x, ln3_g, ln3_b, h3_bf);

  // 8) gate / softmax / top-8  ->  probs (d_out) + gate_dense (ws)
  gate_kernel<<<Tsz, 256, 0, stream>>>(h3_bf, gate_w, task_id, out_prob,
                                       gate_dense);

  // 9) aux loss (deterministic two-stage reduction)
  meanp_kernel<<<Esz, 256, 0, stream>>>(out_prob, mp);
  aux_kernel<<<1, 32, 0, stream>>>(mp, out_aux);

  // 10) x += gate_dense @ b2
  gbias_kernel<<<Tsz, 256, 0, stream>>>(gate_dense, b2, out_x);

  // 11) x += MoE expert mixture
  moe_kernel<<<Tsz / 16, 256, 0, stream>>>(h3_bf, w1t_bf, b1, w2t_bf,
                                           gate_dense, out_x);
}
